// MultiHeadAttention_41936060678646
// MI455X (gfx1250) — matrix operand order repaired and checked by reading, not compiled, not run on hardware
//
#include <hip/hip_runtime.h>

// ---------------------------------------------------------------------------
// MHA block for MI455X (gfx1250, wave32).  v_wmma_f32_16x16x32_bf16 everywhere,
// fp32 accumulate.  CDNA5 data movement: global_load_async_to_lds_b128 for the
// attention K tile (ASYNCcnt), TDM tensor_load_to_lds for the bf16 A tile of
// the output projection (TENSORcnt), global_prefetch for the fp32 GEMM A tiles.
// ---------------------------------------------------------------------------

typedef __attribute__((ext_vector_type(16))) __bf16 v16bf;
typedef __attribute__((ext_vector_type(8)))  __bf16 v8bf;
typedef __attribute__((ext_vector_type(8)))  float  v8f;

union Frag16 { v16bf v; v8bf h[2]; };

// ---------------------------------------------------------------------------
// TDM: 2-D tile of 2-byte elements, global -> LDS.  Builds a D# per ISA 8.3/8.4
// and issues TENSOR_LOAD_TO_LDS.  Guarded: falls back to nothing (caller uses
// the manual path) if the builtin is absent on this toolchain.
// ---------------------------------------------------------------------------
#if __has_builtin(__builtin_amdgcn_tensor_load_to_lds)
#define HAVE_TDM 1
#else
#define HAVE_TDM 0
#endif

__device__ __forceinline__ void tdm_load_2d_b16(unsigned lds_byte, const void* gptr,
                                                unsigned tile_d0, unsigned tile_d1,
                                                unsigned tensor_d0, unsigned tensor_d1,
                                                unsigned long long stride_d0_elems)
{
#if HAVE_TDM
    typedef unsigned int tdm_v4ui __attribute__((ext_vector_type(4)));
    typedef int          tdm_v8i  __attribute__((ext_vector_type(8)));
    typedef int          tdm_v4i  __attribute__((ext_vector_type(4)));
    unsigned long long ga = (unsigned long long)(uintptr_t)gptr;

    tdm_v4ui g0;
    g0.x = 1u;                                        // count=1, user descriptor
    g0.y = lds_byte;                                  // lds_addr [63:32]
    g0.z = (unsigned)(ga & 0xffffffffull);            // global_addr lo
    g0.w = (unsigned)((ga >> 32) & 0x1ffffffull)      // global_addr hi (57b)
         | (2u << 30);                                // type=2 ("image")

    tdm_v8i g1;
    g1[0] = (int)(1u << 16);                          // wg_mask=0, data_size=1 (2B)
    g1[1] = (int)((tensor_d0 & 0xffffu) << 16);       // tensor_dim0[15:0] @ [63:48]
    g1[2] = (int)(((tensor_d0 >> 16) & 0xffffu)       // tensor_dim0[31:16]
                 | ((tensor_d1 & 0xffffu) << 16));    // tensor_dim1[15:0]
    g1[3] = (int)(((tensor_d1 >> 16) & 0xffffu)       // tensor_dim1[31:16]
                 | ((tile_d0 & 0xffffu) << 16));      // tile_dim0
    g1[4] = (int)(tile_d1 & 0xffffu);                 // tile_dim1, tile_dim2=0
    g1[5] = (int)(unsigned)(stride_d0_elems & 0xffffffffull);   // dim0 stride lo
    g1[6] = (int)(unsigned)((stride_d0_elems >> 32) & 0xffffull);
    g1[7] = 0;                                        // dim1 stride = 0 (2-D)

    tdm_v4i z4 = {0, 0, 0, 0};
#if defined(__clang_major__) && __clang_major__ >= 23
    tdm_v8i z8 = {0, 0, 0, 0, 0, 0, 0, 0};
    __builtin_amdgcn_tensor_load_to_lds(g0, g1, z4, z4, z8, 0);
#else
    __builtin_amdgcn_tensor_load_to_lds(g0, g1, z4, z4, 0);
#endif
#endif
}

__device__ __forceinline__ void wait_tensorcnt0()
{
#if __has_builtin(__builtin_amdgcn_s_wait_tensorcnt)
    __builtin_amdgcn_s_wait_tensorcnt(0);
#else
    asm volatile("s_wait_tensorcnt 0" ::: "memory");
#endif
}

// ---------------------------------------------------------------------------
// GEMM: C[M,N] = relu(A[M,K] @ W[K,N] + bias)
// Block tile 128(M) x 64(N), K-step 32, 256 threads = 8 waves.
// bf16-A instantiation stages the A tile via TDM (one DMA per K-step).
// ---------------------------------------------------------------------------
template <typename AT, bool HEAD_OUT>
__global__ __launch_bounds__(256)
void gemm_relu_kernel(const AT* __restrict__ A, const float* __restrict__ W,
                      const float* __restrict__ bias, void* __restrict__ Out,
                      int M, int K, int N, int S, int H, int dh)
{
    constexpr int TM = 128, TN = 64, TK = 32;
    __shared__ __bf16 As[TM][TK];   // 8 KB (TDM destination for bf16 A)
    __shared__ __bf16 Bs[TN][TK];   // 4 KB, transposed: Bs[n][k]

    const int tid    = threadIdx.x;
    const int w      = tid >> 5;
    const int lane   = tid & 31;
    const int laneL  = lane & 15;
    const int laneHi = lane >> 4;
    const int m0 = blockIdx.x * TM;
    const int n0 = blockIdx.y * TN;

    constexpr bool USE_TDM = (HAVE_TDM != 0) && (sizeof(AT) == 2);

    v8f acc[4] = {};

    for (int k0 = 0; k0 < K; k0 += TK) {
        // ---- stage A tile ----
        if constexpr (USE_TDM) {
            if (w == 0) {   // one TDM DMA per workgroup: 128 x 32 bf16, stride K
                tdm_load_2d_b16((unsigned)(uintptr_t)&As[0][0],
                                (const void*)(A + (size_t)m0 * K + k0),
                                TK, TM, (unsigned)K, (unsigned)M,
                                (unsigned long long)K);
            }
        } else {
            for (int idx = tid; idx < TM * TK; idx += 256) {
                int i = idx >> 5, j = idx & 31;
                As[i][j] = (__bf16)(float)A[(size_t)(m0 + i) * K + k0 + j];
            }
        }
        // ---- stage W tile transposed (fp32 -> bf16, coalesced along n) ----
        for (int idx = tid; idx < TK * TN; idx += 256) {
            int kk = idx >> 6, nn = idx & 63;
            Bs[nn][kk] = (__bf16)W[(size_t)(k0 + kk) * N + n0 + nn];
        }
        if constexpr (USE_TDM) { if (w == 0) wait_tensorcnt0(); }
        __syncthreads();

        if (k0 + TK < K) {  // CDNA5 global_prefetch on next A tile
            const char* pf = (const char*)&A[(size_t)(m0 + (tid >> 1)) * K + k0 + TK];
            __builtin_prefetch(pf, 0, 1);
        }

        // preload all fragments, then issue the WMMA chain back-to-back
        Frag16 a;
        const __bf16* ap = &As[w * 16 + laneL][laneHi * 8];
        a.h[0] = *(const v8bf*)ap;
        a.h[1] = *(const v8bf*)(ap + 16);

        Frag16 b[4];
#pragma unroll
        for (int s = 0; s < 4; ++s) {
            const __bf16* bp = &Bs[s * 16 + laneL][laneHi * 8];
            b[s].h[0] = *(const v8bf*)bp;
            b[s].h[1] = *(const v8bf*)(bp + 16);
        }
#pragma unroll
        for (int s = 0; s < 4; ++s)
            acc[s] = __builtin_amdgcn_wmma_f32_16x16x32_bf16(
                false, a.v, false, b[s].v, (short)0, acc[s], false, false);
        __syncthreads();
    }

    // epilogue: C layout = VGPR r -> row r (lanes 0-15) / r+8 (lanes 16-31)
#pragma unroll
    for (int s = 0; s < 4; ++s) {
#pragma unroll
        for (int r = 0; r < 8; ++r) {
            int row = m0 + w * 16 + r + laneHi * 8;
            int col = n0 + s * 16 + laneL;
            float v = acc[s][r] + bias[col];
            v = v > 0.0f ? v : 0.0f;
            if constexpr (HEAD_OUT) {
                int bb = row / S, si = row % S;
                int hh = col / dh, d = col % dh;
                ((__bf16*)Out)[(((size_t)(bb * H + hh)) * S + si) * dh + d] = (__bf16)v;
            } else {
                ((float*)Out)[(size_t)row * N + col] = v;
            }
        }
    }
}

// ---------------------------------------------------------------------------
// Flash attention, causal.  Block = 64 query rows (4 waves x 16), KV tile 64.
// K tile staged with global_load_async_to_lds_b128 (ASYNCcnt); V transposed
// manually; P re-laid-out through per-wave LDS for the PV A-fragments.
// ---------------------------------------------------------------------------
__global__ __launch_bounds__(128)
void flash_attn_kernel(const __bf16* __restrict__ Q, const __bf16* __restrict__ Kb,
                       const __bf16* __restrict__ V, __bf16* __restrict__ Out,
                       int S, int H, int Dtot)
{
    constexpr int DH = 64;
    __shared__ __bf16 Ks[64][DH];        // 8 KB, [key][d], async-LDS destination
    __shared__ __bf16 Vt[DH][64];        // 8 KB, transposed [d][key]
    __shared__ __bf16 Ps[4][16][64];     // 8 KB, per-wave P staging

    const int tid    = threadIdx.x;
    const int w      = tid >> 5;
    const int lane   = tid & 31;
    const int laneL  = lane & 15;
    const int laneHi = lane >> 4;
    const int qblk = blockIdx.x;
    const int h    = blockIdx.y;
    const int b    = blockIdx.z;
    const size_t bh = (size_t)b * H + h;
    const __bf16* Qh = Q  + bh * (size_t)S * DH;
    const __bf16* Kh = Kb + bh * (size_t)S * DH;
    const __bf16* Vh = V  + bh * (size_t)S * DH;

    const int qbase = qblk * 64 + w * 16;

    // Q A-fragments (two dh k-steps), held in registers for the whole kernel
    Frag16 qa[2];
#pragma unroll
    for (int ks = 0; ks < 2; ++ks) {
        const __bf16* qp = Qh + (size_t)(qbase + laneL) * DH + ks * 32 + laneHi * 8;
        qa[ks].h[0] = *(const v8bf*)qp;
        qa[ks].h[1] = *(const v8bf*)(qp + 16);
    }

    v8f o[4] = {};
    float mrow[8], lrow[8];
#pragma unroll
    for (int r = 0; r < 8; ++r) { mrow[r] = -3.0e38f; lrow[r] = 0.0f; }

    const unsigned ksLds = (unsigned)(uintptr_t)&Ks[0][0];

    const int kvEnd = qblk * 64 + 64;
    for (int kv = 0; kv < kvEnd; kv += 64) {
        // ---- K tile: 8 KB contiguous, async copy to LDS (16 B per lane/round)
        {
            const char* gK = (const char*)(Kh + (size_t)kv * DH);
#pragma unroll
            for (int rr = 0; rr < 4; ++rr) {
                unsigned off = (unsigned)(tid * 16 + rr * 2048);
                unsigned la = ksLds + off;
                unsigned long long ga = (unsigned long long)(uintptr_t)(gK + off);
                asm volatile("global_load_async_to_lds_b128 %0, %1, off"
                             :: "v"(la), "v"(ga) : "memory");
            }
        }
        // ---- V tile: load 8-elem chunks, transpose into Vt ----
        for (int idx = tid; idx < 512; idx += 128) {
            int rowk = idx >> 3;
            int c8   = (idx & 7) * 8;
            v8bf vv = *(const v8bf*)(Vh + (size_t)(kv + rowk) * DH + c8);
#pragma unroll
            for (int e = 0; e < 8; ++e) Vt[c8 + e][rowk] = vv[e];
        }
        asm volatile("s_wait_asynccnt 0" ::: "memory");
        __syncthreads();

        // ---- scores S = Q @ K^T : preload 8 B-frags, then 8 WMMAs ----
        Frag16 kf[4][2];
#pragma unroll
        for (int ksub = 0; ksub < 4; ++ksub) {
#pragma unroll
            for (int ks = 0; ks < 2; ++ks) {
                const __bf16* bp = &Ks[ksub * 16 + laneL][ks * 32 + laneHi * 8];
                kf[ksub][ks].h[0] = *(const v8bf*)bp;
                kf[ksub][ks].h[1] = *(const v8bf*)(bp + 16);
            }
        }
        v8f sc[4] = {};
#pragma unroll
        for (int ksub = 0; ksub < 4; ++ksub)
#pragma unroll
            for (int ks = 0; ks < 2; ++ks)
                sc[ksub] = __builtin_amdgcn_wmma_f32_16x16x32_bf16(
                    false, qa[ks].v, false, kf[ksub][ks].v, (short)0,
                    sc[ksub], false, false);

        // ---- online softmax per accumulator register r (row = r+8*laneHi) ----
#pragma unroll
        for (int r = 0; r < 8; ++r) {
            const int qi = qbase + r + laneHi * 8;
            float pv[4];
            float rmax = -3.0e38f;
#pragma unroll
            for (int ksub = 0; ksub < 4; ++ksub) {
                int kj = kv + ksub * 16 + laneL;
                float sv = sc[ksub][r] * 0.125f;       // 1/sqrt(64)
                sv = (kj <= qi) ? sv : -3.0e38f;       // causal mask
                pv[ksub] = sv;
                rmax = fmaxf(rmax, sv);
            }
#pragma unroll
            for (int d = 1; d < 16; d <<= 1)
                rmax = fmaxf(rmax, __shfl_xor(rmax, d, 32));
            float mnew = fmaxf(mrow[r], rmax);
            float corr = __expf(mrow[r] - mnew);
            mrow[r] = mnew;
            float rsum = 0.0f;
#pragma unroll
            for (int ksub = 0; ksub < 4; ++ksub) {
                float p = __expf(pv[ksub] - mnew);
                rsum += p;
                Ps[w][r + laneHi * 8][ksub * 16 + laneL] = (__bf16)p;
            }
#pragma unroll
            for (int d = 1; d < 16; d <<= 1)
                rsum += __shfl_xor(rsum, d, 32);
            lrow[r] = lrow[r] * corr + rsum;
#pragma unroll
            for (int dsub = 0; dsub < 4; ++dsub) o[dsub][r] *= corr;
        }

        // ---- O += P @ V : preload frags, then 8 WMMAs ----
        Frag16 pa[2], vf[4][2];
#pragma unroll
        for (int ks = 0; ks < 2; ++ks) {
            const __bf16* pp = &Ps[w][laneL][ks * 32 + laneHi * 8];
            pa[ks].h[0] = *(const v8bf*)pp;
            pa[ks].h[1] = *(const v8bf*)(pp + 16);
#pragma unroll
            for (int dsub = 0; dsub < 4; ++dsub) {
                const __bf16* vp = &Vt[dsub * 16 + laneL][ks * 32 + laneHi * 8];
                vf[dsub][ks].h[0] = *(const v8bf*)vp;
                vf[dsub][ks].h[1] = *(const v8bf*)(vp + 16);
            }
        }
#pragma unroll
        for (int ks = 0; ks < 2; ++ks)
#pragma unroll
            for (int dsub = 0; dsub < 4; ++dsub)
                o[dsub] = __builtin_amdgcn_wmma_f32_16x16x32_bf16(
                    false, pa[ks].v, false, vf[dsub][ks].v, (short)0,
                    o[dsub], false, false);
        __syncthreads();
    }

    // ---- write bf16 [B,S,D], D-column = h*64 + d ----
#pragma unroll
    for (int dsub = 0; dsub < 4; ++dsub) {
#pragma unroll
        for (int r = 0; r < 8; ++r) {
            int qi = qbase + r + laneHi * 8;
            float val = o[dsub][r] / lrow[r];
            Out[((size_t)b * S + qi) * Dtot + h * 64 + dsub * 16 + laneL] = (__bf16)val;
        }
    }
}

// ---------------------------------------------------------------------------
extern "C" void kernel_launch(void* const* d_in, const int* in_sizes, int n_in,
                              void* d_out, int out_size, void* d_ws, size_t ws_size,
                              hipStream_t stream) {
    const float* query = (const float*)d_in[0];
    const float* key   = (const float*)d_in[1];
    const float* value = (const float*)d_in[2];
    const float* Wq = (const float*)d_in[3];  const float* bq = (const float*)d_in[4];
    const float* Wk = (const float*)d_in[5];  const float* bk = (const float*)d_in[6];
    const float* Wv = (const float*)d_in[7];  const float* bv = (const float*)d_in[8];
    const float* Wo = (const float*)d_in[9];  const float* bo = (const float*)d_in[10];

    const int B = 2, S = 2048, D = 1024, H = 16, dh = 64;
    const int M = B * S;                    // 4096

    __bf16* ws = (__bf16*)d_ws;
    const size_t per = (size_t)M * D;       // 4M bf16 = 8 MB each
    __bf16* Qb = ws;                        // [B,H,S,dh]
    __bf16* Kb = ws + per;
    __bf16* Vb = ws + 2 * per;
    __bf16* Ab = ws + 3 * per;              // attention out, [B,S,D]

    dim3 gg(M / 128, D / 64, 1);
    dim3 gb(256);
    gemm_relu_kernel<float, true><<<gg, gb, 0, stream>>>(query, Wq, bq, Qb, M, D, D, S, H, dh);
    gemm_relu_kernel<float, true><<<gg, gb, 0, stream>>>(key,   Wk, bk, Kb, M, D, D, S, H, dh);
    gemm_relu_kernel<float, true><<<gg, gb, 0, stream>>>(value, Wv, bv, Vb, M, D, D, S, H, dh);

    dim3 ga(S / 64, H, B);
    flash_attn_kernel<<<ga, dim3(128), 0, stream>>>(Qb, Kb, Vb, Ab, S, H, D);

    gemm_relu_kernel<__bf16, false><<<gg, gb, 0, stream>>>(Ab, Wo, bo, d_out, M, D, D, S, H, dh);
}
